// HCNLayerSized_10866267259114
// MI455X (gfx1250) — matrix operand order repaired and checked by reading, not compiled, not run on hardware
//
#include <hip/hip_runtime.h>
#include <stdint.h>

#define N_NODES 50000
#define D_FEAT  256
#define E_DOWN  300000
#define E_UP    100000
#define BN_EPS  1e-5f

typedef __attribute__((ext_vector_type(16))) __bf16 v16bf;
typedef __attribute__((ext_vector_type(8)))  float  v8f;

union FragU {
    v16bf v;
    uint4 q[2];
};

__device__ __forceinline__ unsigned short f32_to_bf16(float f) {
    union { float f; uint32_t u; } c; c.f = f;
    uint32_t u = c.u;
    uint32_t r = u + 0x7FFFu + ((u >> 16) & 1u);   // round-to-nearest-even
    return (unsigned short)(r >> 16);
}

__device__ __forceinline__ unsigned int pack_bf16(float a, float b) {
    return (unsigned int)f32_to_bf16(a) | ((unsigned int)f32_to_bf16(b) << 16);
}

// ---------------------------------------------------------------- utilities
__global__ void k_zero(float* __restrict__ p, long long n) {
    long long i = (long long)blockIdx.x * blockDim.x + threadIdx.x;
    long long stride = (long long)gridDim.x * blockDim.x;
    for (; i < n; i += stride) p[i] = 0.0f;
}

// --------------------------------------------------- edge message + scatter
// m = relu(x[src] + ea); aggr[dst] += m   (float4 per work unit, 64 units/edge)
__global__ void k_scatter(const float* __restrict__ x,
                          const int*   __restrict__ ei,   // [2,E] flat: src then dst
                          const float* __restrict__ ea,
                          float*       __restrict__ aggr,
                          int E) {
    long long total = (long long)E * 64;
    long long i = (long long)blockIdx.x * blockDim.x + threadIdx.x;
    long long stride = (long long)gridDim.x * blockDim.x;
    const float4* x4  = (const float4*)x;
    const float4* ea4 = (const float4*)ea;
    for (; i < total; i += stride) {
        int e = (int)(i >> 6);
        int q = (int)(i & 63);
        int src = ei[e];
        int dst = ei[E + e];
        float4 xv = x4[(long long)src * 64 + q];
        float4 av = ea4[(long long)e * 64 + q];
        float m0 = fmaxf(xv.x + av.x, 0.0f);
        float m1 = fmaxf(xv.y + av.y, 0.0f);
        float m2 = fmaxf(xv.z + av.z, 0.0f);
        float m3 = fmaxf(xv.w + av.w, 0.0f);
        float* out = aggr + (long long)dst * 256 + q * 4;
        atomicAdd(out + 0, m0);
        atomicAdd(out + 1, m1);
        atomicAdd(out + 2, m2);
        atomicAdd(out + 3, m3);
    }
}

// ------------------------------------------------------------ WMMA GEMM
// Y[N,256] = f(A)[N,256] @ W[256,256] + bias
// MODE 1: f(A) = (1+eps)*A + A2         (GINE update input)
// MODE 2: f(A) = relu(BN(A))            (MLP hidden; BN params precomputed/blk)
// Block: 256 thr = 8 waves; block tile 128 rows x 128 cols; K-chunks of 32.
// LDS holds tiles in *fragment-ready* order: per (tile, lane) 16 contiguous
// bf16 (48B-stride slots, 16B aligned) so a fragment = 2x ds_load_b128.
// Out-of-range A rows are clamped (not zeroed): row r of A only feeds output
// row r, and the epilogue never stores out-of-range rows, so clamped loads
// are harmless and the staging loop stays branch-free.
#define FRAG_SLOT 24          // ushorts per lane slot (16 used + 8 pad)
#define TILE_SZ   (32 * FRAG_SLOT)   // 768 ushorts per 16x32 (or 32x16) tile
#define B_OFF     (8 * TILE_SZ)      // B fragments after 8 A tiles

template<int MODE>
__global__ __launch_bounds__(256) void k_gemm(
    const float* __restrict__ A,
    const float* __restrict__ A2,     // MODE 1
    const float* __restrict__ epsP,   // MODE 1 (scalar)
    const float* __restrict__ statS,  // MODE 2: per-col sum
    const float* __restrict__ statQ,  // MODE 2: per-col sumsq
    const float* __restrict__ gam,    // MODE 2
    const float* __restrict__ bet,    // MODE 2
    const float* __restrict__ W,      // [256,256] row-major (k, n)
    const float* __restrict__ bias,   // [256]
    float*       __restrict__ Y,
    int n_rows)
{
    __shared__ unsigned short sLDS[16 * TILE_SZ];   // 24576 B
    __shared__ float sMu[256], sSc[256], sBt[256];  // MODE 2 BN params

    const int tid  = threadIdx.x;
    const int lane = tid & 31;
    const int wave = tid >> 5;        // 0..7 -> 16-row strip
    const int half = lane >> 4;       // 0/1
    const int l16  = lane & 15;
    const int row0 = blockIdx.y * 128;
    const int colB = blockIdx.x * 128;

    float ep1 = 0.0f;
    if (MODE == 1) ep1 = 1.0f + epsP[0];
    if (MODE == 2) {
        const float invN = 1.0f / (float)N_NODES;
        int c = tid;                  // 256 threads == 256 columns
        float mu  = statS[c] * invN;
        float var = statQ[c] * invN - mu * mu;
        sMu[c] = mu;
        sSc[c] = gam[c] * rsqrtf(var + BN_EPS);
        sBt[c] = bet[c];
        __syncthreads();
    }

    v8f acc[8] = {};

    for (int k0 = 0; k0 < 256; k0 += 32) {
        // ---- stage A chunk (128 rows x 32 k) as bf16 pairs, fragment order
        for (int p = tid; p < 2048; p += 256) {
            int r  = p >> 4;
            int k  = (p & 15) << 1;           // even k
            int grow = min(row0 + r, n_rows - 1);   // clamp: branch-free, harmless
            long long idx = (long long)grow * 256 + k0 + k;
            float2 xv = *reinterpret_cast<const float2*>(&A[idx]);
            float v0 = xv.x, v1 = xv.y;
            if (MODE == 1) {
                float2 av = *reinterpret_cast<const float2*>(&A2[idx]);
                v0 = ep1 * v0 + av.x;
                v1 = ep1 * v1 + av.y;
            } else if (MODE == 2) {
                int c = k0 + k;
                v0 = fmaxf((v0 - sMu[c]) * sSc[c] + sBt[c], 0.0f);
                v1 = fmaxf((v1 - sMu[c + 1]) * sSc[c + 1] + sBt[c + 1], 0.0f);
            }
            // A frag layout: half=(k>>3)&1, e=(k&7)+8*(k>>4), lane=m+16*half
            int h  = (k >> 3) & 1;
            int e  = (k & 7) + ((k >> 4) << 3);
            int ln = (r & 15) + 16 * h;
            int ti = r >> 4;
            *reinterpret_cast<unsigned int*>(&sLDS[ti * TILE_SZ + ln * FRAG_SLOT + e]) =
                pack_bf16(v0, v1);
        }
        // ---- stage B chunk (32 k x 128 n) as bf16 pairs, fragment order
        for (int p = tid; p < 2048; p += 256) {
            int n = p & 127;
            int k = (p >> 7) << 1;            // even k
            long long gi = (long long)(k0 + k) * 256 + colB + n;
            float w0 = W[gi];
            float w1 = W[gi + 256];
            // B frag layout: half=(k>>3)&1, e=(k&7)+8*(k>>4), lane=(n&15)+16*half
            int h  = (k >> 3) & 1;
            int e  = (k & 7) + ((k >> 4) << 3);
            int ln = (n & 15) + 16 * h;
            int fr = n >> 4;
            *reinterpret_cast<unsigned int*>(&sLDS[B_OFF + fr * TILE_SZ + ln * FRAG_SLOT + e]) =
                pack_bf16(w0, w1);
        }
        __syncthreads();

        // ---- fragments: 2x ds_load_b128 each; A frag reused across 8 WMMAs.
        FragU fa;
        {
            const uint4* ap =
                reinterpret_cast<const uint4*>(&sLDS[wave * TILE_SZ + lane * FRAG_SLOT]);
            fa.q[0] = ap[0];
            fa.q[1] = ap[1];
        }
        FragU fb[2];
        {
            const uint4* bp =
                reinterpret_cast<const uint4*>(&sLDS[B_OFF + lane * FRAG_SLOT]);
            fb[0].q[0] = bp[0];
            fb[0].q[1] = bp[1];
        }
#pragma unroll
        for (int j = 0; j < 8; ++j) {
            if (j < 7) {
                const uint4* bp = reinterpret_cast<const uint4*>(
                    &sLDS[B_OFF + (j + 1) * TILE_SZ + lane * FRAG_SLOT]);
                fb[(j + 1) & 1].q[0] = bp[0];
                fb[(j + 1) & 1].q[1] = bp[1];
            }
            acc[j] = __builtin_amdgcn_wmma_f32_16x16x32_bf16(false, fa.v, false, fb[j & 1].v,
                                                             (short)0, acc[j], false, false);
        }
        __syncthreads();
    }

    // ---- epilogue: + bias; uniform fast path when whole 128-row block valid
    const bool full = (row0 + 128 <= n_rows);
#pragma unroll
    for (int j = 0; j < 8; ++j) {
        union { v8f v; float f[8]; } o;
        o.v = acc[j];
        const int col = colB + j * 16 + l16;
        const float bb = bias[col];
        const int rbase = row0 + wave * 16 + 8 * half;
        long long base = (long long)rbase * 256 + col;
        if (full) {
#pragma unroll
            for (int r = 0; r < 8; ++r)
                Y[base + (long long)r * 256] = o.f[r] + bb;
        } else {
#pragma unroll
            for (int r = 0; r < 8; ++r)
                if (rbase + r < n_rows)
                    Y[base + (long long)r * 256] = o.f[r] + bb;
        }
    }
}

// ------------------------------------------- per-column sum / sumsq (for BN)
__global__ __launch_bounds__(256) void k_colstats(const float* __restrict__ H,
                                                  float* __restrict__ s,
                                                  float* __restrict__ q,
                                                  int n) {
    int c = threadIdx.x;            // 256 columns
    float ls = 0.0f, lq = 0.0f;
    for (int r = blockIdx.x; r < n; r += gridDim.x) {
        float v = H[(long long)r * 256 + c];
        ls += v;
        lq += v * v;
    }
    atomicAdd(&s[c], ls);
    atomicAdd(&q[c], lq);
}

// ------------------------- residual combine + column stats of pre-BN output
__global__ __launch_bounds__(256) void k_combine(const float* __restrict__ x,
                                                 const float* __restrict__ hd,
                                                 const float* __restrict__ hu,
                                                 const float* __restrict__ a1P,
                                                 const float* __restrict__ a2P,
                                                 float* __restrict__ outpre,
                                                 float* __restrict__ s,
                                                 float* __restrict__ q,
                                                 int n) {
    int c = threadIdx.x;
    float a1 = a1P[0], a2 = a2P[0];
    float ls = 0.0f, lq = 0.0f;
    for (int r = blockIdx.x; r < n; r += gridDim.x) {
        long long i = (long long)r * 256 + c;
        float v = x[i] + a1 * hd[i] + a2 * hu[i];
        outpre[i] = v;
        ls += v;
        lq += v * v;
    }
    atomicAdd(&s[c], ls);
    atomicAdd(&q[c], lq);
}

// ----------------------------------------------------- final BN + ReLU
__global__ void k_bnrelu(const float* __restrict__ pre,
                         const float* __restrict__ s,
                         const float* __restrict__ q,
                         const float* __restrict__ g,
                         const float* __restrict__ b,
                         float* __restrict__ out,
                         int n) {
    long long total = (long long)n * 256;
    long long i = (long long)blockIdx.x * blockDim.x + threadIdx.x;
    long long stride = (long long)gridDim.x * blockDim.x;
    const float invN = 1.0f / (float)n;
    for (; i < total; i += stride) {
        int c = (int)(i & 255);
        float mu  = s[c] * invN;
        float var = q[c] * invN - mu * mu;
        float v = g[c] * (pre[i] - mu) * rsqrtf(var + BN_EPS) + b[c];
        out[i] = fmaxf(v, 0.0f);
    }
}

// ---------------------------------------------------------------- launcher
extern "C" void kernel_launch(void* const* d_in, const int* in_sizes, int n_in,
                              void* d_out, int out_size, void* d_ws, size_t ws_size,
                              hipStream_t stream) {
    const float* x     = (const float*)d_in[0];
    const int*   ei    = (const int*)  d_in[1];
    const float* ea    = (const float*)d_in[2];
    const int*   vi    = (const int*)  d_in[3];
    const float* ve    = (const float*)d_in[4];
    const float* eps_d = (const float*)d_in[5];
    const float* W1d   = (const float*)d_in[6];
    const float* b1d   = (const float*)d_in[7];
    const float* g1d   = (const float*)d_in[8];
    const float* bt1d  = (const float*)d_in[9];
    const float* W2d   = (const float*)d_in[10];
    const float* b2d   = (const float*)d_in[11];
    const float* eps_u = (const float*)d_in[12];
    const float* W1u   = (const float*)d_in[13];
    const float* b1u   = (const float*)d_in[14];
    const float* g1u   = (const float*)d_in[15];
    const float* bt1u  = (const float*)d_in[16];
    const float* W2u   = (const float*)d_in[17];
    const float* b2u   = (const float*)d_in[18];
    const float* bn_g  = (const float*)d_in[19];
    const float* bn_b  = (const float*)d_in[20];
    const float* a1    = (const float*)d_in[21];
    const float* a2    = (const float*)d_in[22];
    float* out = (float*)d_out;

    const long long NF = (long long)N_NODES * 256;
    float* ws    = (float*)d_ws;
    float* aggr  = ws;             // NF  (reused for both convs)
    float* H1    = ws + NF;        // NF  (hidden; reused; also holds out_pre)
    float* hd    = ws + 2 * NF;    // NF
    float* hu    = ws + 3 * NF;    // NF
    float* stats = ws + 4 * NF;    // 1024 floats: [s_h, q_h, s_o, q_o]

    dim3 gGemm(2, (N_NODES + 127) / 128);

    // ---- conv_down
    k_zero<<<512, 256, 0, stream>>>(aggr, NF);
    k_scatter<<<2048, 256, 0, stream>>>(x, ei, ea, aggr, E_DOWN);
    k_gemm<1><<<gGemm, 256, 0, stream>>>(x, aggr, eps_d, nullptr, nullptr, nullptr, nullptr,
                                         W1d, b1d, H1, N_NODES);
    k_zero<<<1, 256, 0, stream>>>(stats, 512);
    k_colstats<<<256, 256, 0, stream>>>(H1, stats, stats + 256, N_NODES);
    k_gemm<2><<<gGemm, 256, 0, stream>>>(H1, nullptr, nullptr, stats, stats + 256, g1d, bt1d,
                                         W2d, b2d, hd, N_NODES);

    // ---- conv_up
    k_zero<<<512, 256, 0, stream>>>(aggr, NF);
    k_scatter<<<1024, 256, 0, stream>>>(x, vi, ve, aggr, E_UP);
    k_gemm<1><<<gGemm, 256, 0, stream>>>(x, aggr, eps_u, nullptr, nullptr, nullptr, nullptr,
                                         W1u, b1u, H1, N_NODES);
    k_zero<<<1, 256, 0, stream>>>(stats, 512);
    k_colstats<<<256, 256, 0, stream>>>(H1, stats, stats + 256, N_NODES);
    k_gemm<2><<<gGemm, 256, 0, stream>>>(H1, nullptr, nullptr, stats, stats + 256, g1u, bt1u,
                                         W2u, b2u, hu, N_NODES);

    // ---- residual + final BN/ReLU
    k_zero<<<1, 256, 0, stream>>>(stats + 512, 512);
    k_combine<<<256, 256, 0, stream>>>(x, hd, hu, a1, a2, H1, stats + 512, stats + 768, N_NODES);
    k_bnrelu<<<2048, 256, 0, stream>>>(H1, stats + 512, stats + 768, bn_g, bn_b, out, N_NODES);
}
